// DINDCNRanker_52390011076691
// MI455X (gfx1250) — matrix operand order repaired and checked by reading, compiled, not run on hardware
//
#include <hip/hip_runtime.h>
#include <hip/hip_bf16.h>

// ---------------------------------------------------------------------------
// DIN/DCN ranker for MI455X (gfx1250, wave32, WMMA f16 16x16x32)
// ---------------------------------------------------------------------------

typedef __attribute__((ext_vector_type(16))) _Float16 v16h;
typedef __attribute__((ext_vector_type(8)))  float    v8f;

#define BATCH 2048
#define LH    200
#define ED    64
#define VV    100000
#define FEATD 1664
#define D1HN  512
#define D2HN  256
#define D3HN  256

// ---- WMMA helpers ----------------------------------------------------------

__device__ __forceinline__ v8f wmma16(v16h a, v16h b, v8f c) {
  // D = A(16x32 f16) * B(32x16 f16) + C(16x16 f32)
  return __builtin_amdgcn_wmma_f32_16x16x32_f16(
      false, a, false, b, (short)0, c, false, false);
}

// A-matrix fragment from row-major LDS: lane = M row (lane&15),
// K elements: [k0 + kg*8 .. +7] and [k0 + 16 + kg*8 .. +7], kg = lane>>4.
__device__ __forceinline__ v16h frag_a(const _Float16* base, int stride,
                                       int row0, int k0, int lane) {
  const _Float16* p = base + (size_t)(row0 + (lane & 15)) * stride
                    + k0 + ((lane >> 4) << 3);
  v16h f;
#pragma unroll
  for (int i = 0; i < 8; ++i) f[i] = p[i];
#pragma unroll
  for (int i = 0; i < 8; ++i) f[8 + i] = p[16 + i];
  return f;
}

// B-matrix fragment from n-major (transposed-weight) LDS: lane = N col,
// K elements: [k0 + kg*8 .. +7] and [k0 + 16 + kg*8 .. +7], kg = lane>>4.
__device__ __forceinline__ v16h frag_b(const _Float16* base, int stride,
                                       int col0, int k0, int lane) {
  const _Float16* p = base + (size_t)(col0 + (lane & 15)) * stride
                    + k0 + ((lane >> 4) << 3);
  v16h f;
#pragma unroll
  for (int i = 0; i < 8; ++i) f[i] = p[i];
#pragma unroll
  for (int i = 0; i < 8; ++i) f[8 + i] = p[16 + i];
  return f;
}

// ---- block reductions (wave32) ---------------------------------------------

__device__ __forceinline__ float blockReduceSum(float v, float* s, int tid) {
#pragma unroll
  for (int off = 16; off > 0; off >>= 1) v += __shfl_down(v, off, 32);
  if ((tid & 31) == 0) s[tid >> 5] = v;
  __syncthreads();
  if (tid == 0) {
    float r = s[0];
#pragma unroll
    for (int i = 1; i < 8; ++i) r += s[i];
    s[8] = r;
  }
  __syncthreads();
  return s[8];
}

__device__ __forceinline__ float blockReduceMax(float v, float* s, int tid) {
#pragma unroll
  for (int off = 16; off > 0; off >>= 1) v = fmaxf(v, __shfl_down(v, off, 32));
  if ((tid & 31) == 0) s[tid >> 5] = v;
  __syncthreads();
  if (tid == 0) {
    float r = s[0];
#pragma unroll
    for (int i = 1; i < 8; ++i) r = fmaxf(r, s[i]);
    s[8] = r;
  }
  __syncthreads();
  return s[8];
}

// ---- dynamic LDS layout for the fused attention kernel ---------------------
// sAp : A' = [h | q*h]   208 x 136 halfs   (K=128 + pad)
// sW  : layer1 W'^T 256x136 / layer2 A2^T 128x264 (shared region, 34816 halfs)
// sH1 : H1 208 x 264 halfs (N=256 + pad)
// sH2 : H2 208 x 136 halfs (N=128 + pad)
#define SAP_H (208 * 136)
#define SW_H  34816
#define SH1_H (208 * 264)
#define SH2_H (208 * 136)
#define TOT_H (SAP_H + SW_H + SH1_H + SH2_H)
#define SF_F  1104
#define SMEM1_BYTES ((size_t)TOT_H * 2 + (size_t)SF_F * 4)

__launch_bounds__(256)
__global__ void attn_feat_kernel(
    const int* __restrict__ user_cat,  const float* __restrict__ user_num,
    const int* __restrict__ ctx_cat,   const float* __restrict__ ctx_num,
    const int* __restrict__ hist_ids,  const int* __restrict__ hist_author_ids,
    const int* __restrict__ hist_tag_ids, const int* __restrict__ hist_mask,
    const int* __restrict__ item_cat,  const float* __restrict__ item_num,
    const float* __restrict__ user_emb, const float* __restrict__ item_emb,
    const float* __restrict__ ctx_emb,
    const float* __restrict__ Wu, const float* __restrict__ bu,
    const float* __restrict__ Wi, const float* __restrict__ bi,
    const float* __restrict__ Wc, const float* __restrict__ bc,
    const float* __restrict__ A1, const float* __restrict__ a1,
    const float* __restrict__ A2, const float* __restrict__ a2,
    const float* __restrict__ A3, const float* __restrict__ a3,
    float* __restrict__ basef, _Float16* __restrict__ baseh)
{
  extern __shared__ unsigned char smem[];
  _Float16* sAp = (_Float16*)smem;           // 208 x 136
  _Float16* sW  = sAp + SAP_H;               // shared weight region
  _Float16* sH1 = sW + SW_H;                 // 208 x 264
  _Float16* sH2 = sH1 + SH1_H;               // 208 x 136
  float* sF   = (float*)(sH2 + SH2_H);
  float* sQ   = sF;            // 64
  float* sC1  = sQ + 64;       // 256
  float* sSc  = sC1 + 256;     // 256
  float* sA3  = sSc + 256;     // 128
  float* sRed = sA3 + 128;     // 16
  float* sAtt = sRed + 16;     // 64
  float* sAuth= sAtt + 64;     // 64
  float* sTag = sAuth + 64;    // 64
  float* sNum = sTag + 64;     // 192

  const int b   = blockIdx.x;
  const int tid = threadIdx.x;
  const int lane = tid & 31;
  const int wv   = tid >> 5;   // 0..7

  // ---- prologue: target row q ----------------------------------------------
  const int target = item_cat[(size_t)b * 8 + 0];
  if (tid < ED) sQ[tid] = item_emb[((size_t)0 * VV + target) * ED + tid];
  __syncthreads();

  // const1[n] = a1[n] + sum_d q[d]*(A1[d][n] + A1[128+d][n])
  {
    int n = tid;  // 256 threads == 256 n
    float acc = a1[n];
#pragma unroll 4
    for (int d = 0; d < ED; ++d)
      acc += sQ[d] * (A1[(size_t)d * 256 + n] + A1[(size_t)(128 + d) * 256 + n]);
    sC1[n] = acc;
  }

  // gather A' = [h | q*h] (f16), rows >= LH zero-padded
  for (int idx = tid; idx < 208 * ED; idx += 256) {
    int m = idx >> 6, e = idx & 63;
    if (m < LH) {
      int id = hist_ids[(size_t)b * LH + m];
      float h = item_emb[((size_t)0 * VV + id) * ED + e];
      sAp[m * 136 + e]      = (_Float16)h;
      sAp[m * 136 + 64 + e] = (_Float16)(h * sQ[e]);
    } else {
      sAp[m * 136 + e]      = (_Float16)0.f;
      sAp[m * 136 + 64 + e] = (_Float16)0.f;
    }
  }

  // W'^T[n][k]: k<64 -> A1[64+k][n]-A1[128+k][n]; k>=64 -> A1[128+k][n]
  for (int idx = tid; idx < 128 * 256; idx += 256) {
    int k = idx >> 8, n = idx & 255;
    float w = (k < 64)
        ? (A1[(size_t)(64 + k) * 256 + n] - A1[(size_t)(128 + k) * 256 + n])
        : A1[(size_t)(128 + k) * 256 + n];
    sW[n * 136 + k] = (_Float16)w;
  }
  __syncthreads();

  // ---- layer 1: H1 = relu(const1 + A'(208x128) @ W'(128x256)) --------------
  for (int t = wv; t < 13 * 16; t += 8) {
    int mt = t % 13, nt = t / 13;
    v8f acc;
#pragma unroll
    for (int v = 0; v < 8; ++v) acc[v] = 0.f;
#pragma unroll
    for (int k0 = 0; k0 < 128; k0 += 32) {
      v16h aF = frag_a(sAp, 136, mt * 16, k0, lane);
      v16h bF = frag_b(sW, 136, nt * 16, k0, lane);
      acc = wmma16(aF, bF, acc);
    }
    int dm = mt * 16 + ((lane >> 4) << 3);
    int dn = nt * 16 + (lane & 15);
    float bias = sC1[dn];
#pragma unroll
    for (int v = 0; v < 8; ++v) {
      float x = fmaxf(acc[v] + bias, 0.f);
      sH1[(dm + v) * 264 + dn] = (_Float16)x;
    }
  }
  __syncthreads();

  // ---- layer 2 weights: A2^T[n][k] (128 x 264) into sW region --------------
  for (int idx = tid; idx < 256 * 128; idx += 256) {
    int k = idx >> 7, n = idx & 127;
    sW[n * 264 + k] = (_Float16)A2[(size_t)k * 128 + n];
  }
  __syncthreads();

  // ---- layer 2: H2 = relu(H1(208x256) @ A2(256x128) + a2) ------------------
  for (int t = wv; t < 13 * 8; t += 8) {
    int mt = t % 13, nt = t / 13;
    v8f acc;
#pragma unroll
    for (int v = 0; v < 8; ++v) acc[v] = 0.f;
#pragma unroll
    for (int k0 = 0; k0 < 256; k0 += 32) {
      v16h aF = frag_a(sH1, 264, mt * 16, k0, lane);
      v16h bF = frag_b(sW, 264, nt * 16, k0, lane);
      acc = wmma16(aF, bF, acc);
    }
    int dm = mt * 16 + ((lane >> 4) << 3);
    int dn = nt * 16 + (lane & 15);
    float bias = a2[dn];
#pragma unroll
    for (int v = 0; v < 8; ++v) {
      float x = fmaxf(acc[v] + bias, 0.f);
      sH2[(dm + v) * 136 + dn] = (_Float16)x;
    }
  }
  if (tid < 128) sA3[tid] = A3[tid];
  __syncthreads();

  // ---- layer 3 + masked softmax --------------------------------------------
  float sc = -1e9f;
  int mk = 0;
  if (tid < LH) {
    float d = a3[0];
#pragma unroll 4
    for (int k = 0; k < 128; ++k) d += (float)sH2[tid * 136 + k] * sA3[k];
    mk = hist_mask[(size_t)b * LH + tid];
    sc = (mk <= 0) ? -1e9f : d;
  }
  float maxv = blockReduceMax(sc, sRed, tid);
  float ex = (tid < LH && mk > 0) ? __expf(sc - maxv) : 0.f;
  float sm   = blockReduceSum(ex, sRed, tid);
  float msum = blockReduceSum((tid < LH && mk > 0) ? 1.f : 0.f, sRed, tid);
  float w = ex / fmaxf(sm, 1e-6f);
  if (tid < 208) sSc[tid] = (tid < LH) ? w : 0.f;
  __syncthreads();

  // ---- pools + small numeric MLPs (4 groups of 64 threads) -----------------
  {
    int g = tid >> 6, e = tid & 63;
    if (g == 0) {
      float acc = 0.f;
      for (int m = 0; m < LH; ++m) acc += sSc[m] * (float)sAp[m * 136 + e];
      sAtt[e] = acc;
    } else if (g == 1) {
      float acc = 0.f;
      for (int m = 0; m < LH; ++m) {
        if (hist_mask[(size_t)b * LH + m] > 0) {
          int id = hist_author_ids[(size_t)b * LH + m];
          acc += item_emb[((size_t)1 * VV + id) * ED + e];
        }
      }
      sAuth[e] = acc / fmaxf(msum, 1e-6f);
    } else if (g == 2) {
      float acc = 0.f;
      for (int m = 0; m < LH; ++m) {
        if (hist_mask[(size_t)b * LH + m] > 0) {
          int id = hist_tag_ids[(size_t)b * LH + m];
          acc += item_emb[((size_t)7 * VV + id) * ED + e];
        }
      }
      sTag[e] = acc / fmaxf(msum, 1e-6f);
    } else {
      float un = bu[e];
#pragma unroll
      for (int d = 0; d < 8; ++d) un += user_num[(size_t)b * 8 + d] * Wu[d * 64 + e];
      sNum[e] = fmaxf(un, 0.f);
      float im = bi[e];
#pragma unroll
      for (int d = 0; d < 6; ++d) im += item_num[(size_t)b * 6 + d] * Wi[d * 64 + e];
      sNum[64 + e] = fmaxf(im, 0.f);
      float cn = bc[e];
#pragma unroll
      for (int d = 0; d < 4; ++d) cn += ctx_num[(size_t)b * 4 + d] * Wc[d * 64 + e];
      sNum[128 + e] = fmaxf(cn, 0.f);
    }
  }
  __syncthreads();

  // ---- assemble base feature vector (1664) ---------------------------------
  for (int idx = tid; idx < FEATD; idx += 256) {
    float v;
    if (idx < 384) {
      int j = idx >> 6, e = idx & 63;
      int id = user_cat[(size_t)b * 6 + j];
      v = user_emb[((size_t)j * VV + id) * ED + e];
    } else if (idx < 896) {
      int t = idx - 384, j = t >> 6, e = t & 63;
      int id = item_cat[(size_t)b * 8 + j];
      v = item_emb[((size_t)j * VV + id) * ED + e];
    } else if (idx < 1152) {
      int t = idx - 896, j = t >> 6, e = t & 63;
      int id = ctx_cat[(size_t)b * 4 + j];
      v = ctx_emb[((size_t)j * VV + id) * ED + e];
    } else if (idx < 1344) {
      v = sNum[idx - 1152];
    } else if (idx < 1408) {
      v = sAtt[idx - 1344];
    } else if (idx < 1472) {
      v = sAuth[idx - 1408];
    } else if (idx < 1536) {
      v = sTag[idx - 1472];
    } else if (idx < 1600) {
      int e = idx - 1536; v = sQ[e] * sAtt[e];
    } else {
      int e = idx - 1600; v = sQ[e] - sAtt[e];
    }
    basef[(size_t)b * FEATD + idx] = v;
    baseh[(size_t)b * FEATD + idx] = (_Float16)v;
  }
}

// ---------------------------------------------------------------------------
// Generic WMMA GEMM: out = act(A(MxK f16) @ W(KxN f32) + bias)
// 128x128 tile per block, 8 waves x (2x4) 16x16 tiles, K staged 64 at a time.
// ---------------------------------------------------------------------------
__launch_bounds__(256)
__global__ void gemm_bias_kernel(const _Float16* __restrict__ A,
                                 const float* __restrict__ W,
                                 const float* __restrict__ bias,
                                 _Float16* __restrict__ outH,
                                 float* __restrict__ outF,
                                 int M, int N, int K, int do_relu)
{
  __shared__ _Float16 sA[128 * 72];
  __shared__ _Float16 sBT[128 * 72];
  const int tid = threadIdx.x, lane = tid & 31, wv = tid >> 5;
  const int m0 = blockIdx.x * 128, n0 = blockIdx.y * 128;
  const int wm = wv >> 1, wn = wv & 1;

  v8f acc[2][4];
#pragma unroll
  for (int i = 0; i < 2; ++i)
#pragma unroll
    for (int j = 0; j < 4; ++j)
#pragma unroll
      for (int v = 0; v < 8; ++v) acc[i][j][v] = 0.f;

  for (int kc = 0; kc < K; kc += 64) {
    for (int idx = tid; idx < 128 * 64; idx += 256) {
      int r = idx >> 6, c = idx & 63;
      sA[r * 72 + c] = A[(size_t)(m0 + r) * K + kc + c];
    }
    for (int idx = tid; idx < 64 * 128; idx += 256) {
      int k = idx >> 7, n = idx & 127;
      sBT[n * 72 + k] = (_Float16)W[(size_t)(kc + k) * N + n0 + n];
    }
    __syncthreads();
#pragma unroll
    for (int kk = 0; kk < 64; kk += 32) {
      v16h aF[2], bF[4];
#pragma unroll
      for (int i = 0; i < 2; ++i) aF[i] = frag_a(sA, 72, wm * 32 + i * 16, kk, lane);
#pragma unroll
      for (int j = 0; j < 4; ++j) bF[j] = frag_b(sBT, 72, wn * 64 + j * 16, kk, lane);
#pragma unroll
      for (int i = 0; i < 2; ++i)
#pragma unroll
        for (int j = 0; j < 4; ++j) acc[i][j] = wmma16(aF[i], bF[j], acc[i][j]);
    }
    __syncthreads();
  }

#pragma unroll
  for (int i = 0; i < 2; ++i)
#pragma unroll
    for (int j = 0; j < 4; ++j) {
      int dm = m0 + wm * 32 + i * 16 + ((lane >> 4) << 3);
      int dn = n0 + wn * 64 + j * 16 + (lane & 15);
      float bs = bias[dn];
#pragma unroll
      for (int v = 0; v < 8; ++v) {
        float x = acc[i][j][v] + bs;
        if (do_relu) x = fmaxf(x, 0.f);
        if (outH) outH[(size_t)(dm + v) * N + dn] = (_Float16)x;
        if (outF) outF[(size_t)(dm + v) * N + dn] = x;
      }
    }
}

// ---------------------------------------------------------------------------
// Cross network (3 rank-1 updates) + final head.  One block per batch row.
// ---------------------------------------------------------------------------
__launch_bounds__(256)
__global__ void cross_head_kernel(const float* __restrict__ basef,
                                  const float* __restrict__ deep,
                                  const float* __restrict__ cross_w,
                                  const float* __restrict__ cross_b,
                                  const float* __restrict__ Wh,
                                  const float* __restrict__ bh,
                                  float* __restrict__ out)
{
  __shared__ float sB[FEATD];
  __shared__ float sX[FEATD];
  __shared__ float sR[16];
  const int b = blockIdx.x, tid = threadIdx.x;

  for (int i = tid; i < FEATD; i += 256) {
    float v = basef[(size_t)b * FEATD + i];
    sB[i] = v; sX[i] = v;
  }
  __syncthreads();

  for (int k = 0; k < 3; ++k) {
    float p = 0.f;
    for (int i = tid; i < FEATD; i += 256) p += sX[i] * cross_w[(size_t)k * FEATD + i];
    float xw = blockReduceSum(p, sR, tid);
    for (int i = tid; i < FEATD; i += 256)
      sX[i] = sB[i] * xw + cross_b[(size_t)k * FEATD + i] + sX[i];
    __syncthreads();
  }

  float p = 0.f;
  for (int i = tid; i < FEATD; i += 256) p += sX[i] * Wh[i];
  p += deep[(size_t)b * D3HN + tid] * Wh[FEATD + tid];
  float tot = blockReduceSum(p, sR, tid);
  if (tid == 0) out[b] = tot + bh[0];
}

// ---------------------------------------------------------------------------
// launch
// ---------------------------------------------------------------------------
extern "C" void kernel_launch(void* const* d_in, const int* in_sizes, int n_in,
                              void* d_out, int out_size, void* d_ws, size_t ws_size,
                              hipStream_t stream) {
  (void)in_sizes; (void)n_in; (void)out_size; (void)ws_size;
  const int*   user_cat        = (const int*)  d_in[0];
  const float* user_num        = (const float*)d_in[1];
  const int*   ctx_cat         = (const int*)  d_in[2];
  const float* ctx_num         = (const float*)d_in[3];
  const int*   hist_ids        = (const int*)  d_in[4];
  const int*   hist_author_ids = (const int*)  d_in[5];
  const int*   hist_tag_ids    = (const int*)  d_in[6];
  const int*   hist_mask       = (const int*)  d_in[7];
  const int*   item_cat        = (const int*)  d_in[8];
  const float* item_num        = (const float*)d_in[9];
  const float* user_emb        = (const float*)d_in[10];
  const float* item_emb        = (const float*)d_in[11];
  const float* ctx_emb         = (const float*)d_in[12];
  const float* Wu = (const float*)d_in[13];  const float* bu = (const float*)d_in[14];
  const float* Wi = (const float*)d_in[15];  const float* bi = (const float*)d_in[16];
  const float* Wc = (const float*)d_in[17];  const float* bc = (const float*)d_in[18];
  const float* A1 = (const float*)d_in[19];  const float* a1 = (const float*)d_in[20];
  const float* A2 = (const float*)d_in[21];  const float* a2 = (const float*)d_in[22];
  const float* A3 = (const float*)d_in[23];  const float* a3 = (const float*)d_in[24];
  const float* cross_w = (const float*)d_in[25];
  const float* cross_b = (const float*)d_in[26];
  const float* D1 = (const float*)d_in[27];  const float* d1 = (const float*)d_in[28];
  const float* D2 = (const float*)d_in[29];  const float* d2 = (const float*)d_in[30];
  const float* D3 = (const float*)d_in[31];  const float* d3 = (const float*)d_in[32];
  const float* Wh = (const float*)d_in[33];  const float* bh = (const float*)d_in[34];

  // workspace layout
  char* ws = (char*)d_ws;
  const size_t BASEF_OFF = 0;
  const size_t BASEH_OFF = BASEF_OFF + (size_t)BATCH * FEATD * 4;  // 13,631,488
  const size_t H1_OFF    = BASEH_OFF + (size_t)BATCH * FEATD * 2;  // 20,447,232
  const size_t H2_OFF    = H1_OFF    + (size_t)BATCH * D1HN * 2;   // 22,544,384
  const size_t DEEP_OFF  = H2_OFF    + (size_t)BATCH * D2HN * 2;   // 23,592,960
  float*     basef = (float*)    (ws + BASEF_OFF);
  _Float16*  baseh = (_Float16*) (ws + BASEH_OFF);
  _Float16*  h1    = (_Float16*) (ws + H1_OFF);
  _Float16*  h2    = (_Float16*) (ws + H2_OFF);
  float*     deep  = (float*)    (ws + DEEP_OFF);
  float*     out   = (float*)d_out;

  (void)hipFuncSetAttribute((const void*)attn_feat_kernel,
                            hipFuncAttributeMaxDynamicSharedMemorySize,
                            (int)SMEM1_BYTES);

  attn_feat_kernel<<<BATCH, 256, SMEM1_BYTES, stream>>>(
      user_cat, user_num, ctx_cat, ctx_num, hist_ids, hist_author_ids,
      hist_tag_ids, hist_mask, item_cat, item_num, user_emb, item_emb, ctx_emb,
      Wu, bu, Wi, bi, Wc, bc, A1, a1, A2, a2, A3, a3, basef, baseh);

  gemm_bias_kernel<<<dim3(BATCH / 128, D1HN / 128), 256, 0, stream>>>(
      baseh, D1, d1, h1, nullptr, BATCH, D1HN, FEATD, 1);
  gemm_bias_kernel<<<dim3(BATCH / 128, D2HN / 128), 256, 0, stream>>>(
      h1, D2, d2, h2, nullptr, BATCH, D2HN, D1HN, 1);
  gemm_bias_kernel<<<dim3(BATCH / 128, D3HN / 128), 256, 0, stream>>>(
      h2, D3, d3, nullptr, deep, BATCH, D3HN, D2HN, 0);

  cross_head_kernel<<<BATCH, 256, 0, stream>>>(
      basef, deep, cross_w, cross_b, Wh, bh, out);
}
